// ForwardBackwardGNN_71021579206886
// MI455X (gfx1250) — compile-verified, matrix operand order checked by reading, not run on hardware
//
#include <hip/hip_runtime.h>
#include <math.h>
#include <float.h>

typedef _Float16 half_t;
typedef __attribute__((ext_vector_type(16))) _Float16 v16h;
typedef __attribute__((ext_vector_type(8)))  _Float16 v8h;
typedef __attribute__((ext_vector_type(8)))  float    v8f;

#define N_NODES    50000
#define N_EDGES    400000
#define LSEQ       16
#define EMB_DIM    32
#define HDIM       128
#define DDIM       373
#define DPAD       384
#define SID_N      53
#define IN_IDX_C   161
#define OUT_IDX_C  320
#define OUT_STRIDE (2*DDIM)
#define NSLOPE     0.2f

// weight image staged to LDS: embed(4096) BxF(32768) BhF(131072) BxR(32768)
// BW1(16384) bf(2048) br(2048)  = 221184 bytes, contiguous in workspace
#define WIMG_BYTES 221184
#define LSTM_SMEM  (WIMG_BYTES + 4 * (4096 + 8192))   // + per-wave h(f16) and c(f32)

static __device__ __forceinline__ float sigmoidf_(float x) { return 1.f / (1.f + __expf(-x)); }
static __device__ __forceinline__ float tanhf_(float x)    { return 1.f - 2.f / (__expf(2.f * x) + 1.f); }

static __device__ __forceinline__ v8f wmma16(v16h a, v16h b, v8f c) {
  // v_wmma_f32_16x16x32_f16  (neg_a, A, neg_b, B, c_mod, C, reuse_a, reuse_b)
  return __builtin_amdgcn_wmma_f32_16x16x32_f16(false, a, false, b, (short)0, c, false, false);
}

// A-operand (16x32 f16) lane layout per CDNA5 ISA 7.12.2:
// lane L holds row M=L%16; half j -> K = (L<16?0:8) + (j<8 ? j : j+8)
static __device__ __forceinline__ v16h load_a16(const half_t* rowk, int koff) {
  union { v16h v; v8h h[2]; } u;
  u.h[0] = *(const v8h*)(rowk + koff);        // K = koff+0..7
  u.h[1] = *(const v8h*)(rowk + koff + 16);   // K = koff+16..23
  return u.v;
}

// B-operand pre-packed so each lane's 16 halves are contiguous (32B per lane, 1KB per tile)
static __device__ __forceinline__ v16h load_bpack(const half_t* tile, int lane) {
  return *(const v16h*)(tile + lane * 16);
}

static __device__ __forceinline__ void atomic_max_f32(float* addr, float val) {
  int* ai = (int*)addr;
  int cur = __float_as_int(*addr);
  while (__int_as_float(cur) < val) {
    int old = atomicCAS(ai, cur, __float_as_int(val));
    if (old == cur) break;
    cur = old;
  }
}

// ---------------------------------------------------------------- utility
__global__ void k_fill_f32(float* p, long long n, float v) {
  long long i = (long long)blockIdx.x * blockDim.x + threadIdx.x;
  long long s = (long long)gridDim.x * blockDim.x;
  for (; i < n; i += s) p[i] = v;
}

__global__ void k_copy_f32(const float* __restrict__ src, float* __restrict__ dst, long long n) {
  long long i = (long long)blockIdx.x * blockDim.x + threadIdx.x;
  long long s = (long long)gridDim.x * blockDim.x;
  for (; i < n; i += s) dst[i] = src[i];
}

// embed row 0 zero (padding_idx), combined LSTM biases
__global__ void k_prep_misc(const float* __restrict__ embed,
                            const float* __restrict__ bihf, const float* __restrict__ bhhf,
                            const float* __restrict__ bihr, const float* __restrict__ bhhr,
                            half_t* __restrict__ embed_h, float* __restrict__ bf, float* __restrict__ br) {
  int i = blockIdx.x * blockDim.x + threadIdx.x;
  if (i < 64 * EMB_DIM) embed_h[i] = (half_t)((i < EMB_DIM) ? 0.f : embed[i]);
  if (i < 4 * HDIM) { bf[i] = bihf[i] + bhhf[i]; br[i] = bihr[i] + bhhr[i]; }
}

// Pack a [K,N] (or transposed [N,K]) f32 matrix into WMMA-B lane-order f16 tiles.
// out tile index = nb*kTiles + kb ; within tile: lane(32) x half(16),
// K = kb*32 + (lane<16?0:8) + (j<8 ? j : j+8), Ncol = nb*16 + (lane%16)
__global__ void k_pack_b(const float* __restrict__ W, half_t* __restrict__ out,
                         int K, int N, int ldw, int trans, int kTiles, int nTiles) {
  int total = kTiles * nTiles * 512;
  for (int idx = blockIdx.x * blockDim.x + threadIdx.x; idx < total; idx += gridDim.x * blockDim.x) {
    int tile = idx >> 9;
    int within = idx & 511;
    int lane = within >> 4;
    int j = within & 15;
    int nb = tile / kTiles;
    int kb = tile - nb * kTiles;
    int k = kb * 32 + ((lane & 16) ? 8 : 0) + ((j < 8) ? j : j + 8);
    int n = nb * 16 + (lane & 15);
    float v = 0.f;
    if (k < K && n < N) v = trans ? W[(size_t)n * ldw + k] : W[(size_t)k * ldw + n];
    out[idx] = (half_t)v;
  }
}

// ------------------------------------------------------- LSTM edge encoder
// One wave = 16 edges. All LSTM/MLP weights live in LDS (streamed in with
// CDNA5 async global->LDS loads). Cell state c lives in a per-wave LDS tile
// to keep VGPR pressure low (no spills). 160 WMMAs per timestep per wave.
__global__ void k_lstm_edges(const int* __restrict__ tokens,       // [E,16]
                             const half_t* __restrict__ g_wimg,    // contiguous weight image
                             const float* __restrict__ b1, const float* __restrict__ W2,
                             const float* __restrict__ b2, float* __restrict__ ea) {
  extern __shared__ char smem_raw[];
  half_t* s_embed = (half_t*)smem_raw;            // 2048 halves
  half_t* s_BxF = s_embed + 2048;                 // 16384
  half_t* s_BhF = s_BxF + 16384;                  // 65536
  half_t* s_BxR = s_BhF + 65536;                  // 16384
  half_t* s_BW1 = s_BxR + 16384;                  // 8192
  float*  s_bf  = (float*)(s_BW1 + 8192);         // 512
  float*  s_br  = s_bf + 512;                     // 512
  char*   s_wv  = (char*)(s_br + 512);            // per-wave h(4KB f16) + c(8KB f32)

  const int tid = threadIdx.x;
  const int lane = tid & 31;
  const int wave = tid >> 5;
  const int nwaves = blockDim.x >> 5;
  half_t* myH = (half_t*)(s_wv + wave * (4096 + 8192));
  float*  myC = (float*)((char*)myH + 4096);

  // ---- async global->LDS staging of the whole weight image (ASYNCcnt path)
  {
    const char* gsrc = (const char*)g_wimg;
    for (int i = tid; i < (WIMG_BYTES / 16); i += blockDim.x) {
      unsigned lds_off = (unsigned)(i * 16);      // dynamic LDS starts at offset 0
      const char* ga = gsrc + (size_t)i * 16;
      asm volatile("global_load_async_to_lds_b128 %0, %1, off"
                   :: "v"(lds_off), "v"(ga) : "memory");
    }
    asm volatile("s_wait_asynccnt 0x0" ::: "memory");
  }
  __syncthreads();

  const int row  = lane & 15;                 // A row / C col within tile
  const int koff = (lane & 16) ? 8 : 0;       // A-layout K offset
  const int mrow = (lane & 16) ? 8 : 0;       // C-layout row offset

  v8f zacc;
  #pragma unroll
  for (int r = 0; r < 8; ++r) zacc[r] = 0.f;

  const int totalTiles = N_EDGES / 16;
  int wg = blockIdx.x * nwaves + wave;
  int wstride = gridDim.x * nwaves;

  #pragma unroll 1
  for (int tile = wg; tile < totalTiles; tile += wstride) {
    const int e0 = tile * 16;
    const int* tokRow = tokens + (size_t)(e0 + row) * LSEQ;
    v16h ha[4];

    // ---- t = 0 (h = c = 0): gates from x only
    {
      int tok = tokRow[0];
      v16h xa = load_a16(s_embed + tok * EMB_DIM, koff);
      #pragma unroll 1
      for (int nb = 0; nb < 8; ++nb) {
        v8f a0 = wmma16(xa, load_bpack(s_BxF + (0 * 8 + nb) * 512, lane), zacc);
        v8f a1 = wmma16(xa, load_bpack(s_BxF + (1 * 8 + nb) * 512, lane), zacc);
        v8f a2 = wmma16(xa, load_bpack(s_BxF + (2 * 8 + nb) * 512, lane), zacc);
        v8f a3 = wmma16(xa, load_bpack(s_BxF + (3 * 8 + nb) * 512, lane), zacc);
        int col = nb * 16 + row;
        float bi = s_bf[col], bg = s_bf[256 + col], bo = s_bf[384 + col];
        #pragma unroll
        for (int r = 0; r < 8; ++r) {
          float iv = sigmoidf_(a0[r] + bi);
          float gv = tanhf_(a2[r] + bg);
          float ov = sigmoidf_(a3[r] + bo);
          float cc = iv * gv;
          int ci = (mrow + r) * HDIM + col;
          myC[ci] = cc;
          myH[ci] = (half_t)(ov * tanhf_(cc));
        }
        (void)a1;
      }
      #pragma unroll
      for (int kb = 0; kb < 4; ++kb)
        ha[kb] = load_a16(myH + row * HDIM + kb * 32, koff);
    }

    // ---- t = 1..15, branch-free: 5 WMMAs per gate tile
    #pragma unroll 1
    for (int t = 1; t < LSEQ; ++t) {
      int tok = tokRow[t];
      v16h xa = load_a16(s_embed + tok * EMB_DIM, koff);
      #pragma unroll 1
      for (int nb = 0; nb < 8; ++nb) {
        v8f acc[4];
        #pragma unroll
        for (int g = 0; g < 4; ++g) {
          int gt = g * 8 + nb;
          v8f a = wmma16(xa, load_bpack(s_BxF + gt * 512, lane), zacc);
          #pragma unroll
          for (int kb = 0; kb < 4; ++kb)
            a = wmma16(ha[kb], load_bpack(s_BhF + (gt * 4 + kb) * 512, lane), a);
          acc[g] = a;
        }
        int col = nb * 16 + row;
        float bi = s_bf[col], bff = s_bf[128 + col], bg = s_bf[256 + col], bo = s_bf[384 + col];
        #pragma unroll
        for (int r = 0; r < 8; ++r) {
          int ci = (mrow + r) * HDIM + col;
          float iv = sigmoidf_(acc[0][r] + bi);
          float fv = sigmoidf_(acc[1][r] + bff);
          float gv = tanhf_(acc[2][r] + bg);
          float ov = sigmoidf_(acc[3][r] + bo);
          float cc = fv * myC[ci] + iv * gv;
          myC[ci] = cc;
          myH[ci] = (half_t)(ov * tanhf_(cc));
        }
      }
      // transpose h back to A-layout for next step (per-wave LDS, in-order)
      #pragma unroll
      for (int kb = 0; kb < 4; ++kb)
        ha[kb] = load_a16(myH + row * HDIM + kb * 32, koff);
    }

    // ---- backward LSTM: single cell on last token, h0=c0=0
    {
      int tok = tokRow[LSEQ - 1];
      v16h xa = load_a16(s_embed + tok * EMB_DIM, koff);
      #pragma unroll 1
      for (int nb = 0; nb < 8; ++nb) {
        v8f a0 = wmma16(xa, load_bpack(s_BxR + (0 * 8 + nb) * 512, lane), zacc);
        v8f a2 = wmma16(xa, load_bpack(s_BxR + (2 * 8 + nb) * 512, lane), zacc);
        v8f a3 = wmma16(xa, load_bpack(s_BxR + (3 * 8 + nb) * 512, lane), zacc);
        int col = nb * 16 + row;
        float bi = s_br[col], bg = s_br[256 + col], bo = s_br[384 + col];
        #pragma unroll
        for (int r = 0; r < 8; ++r) {
          float iv = sigmoidf_(a0[r] + bi);
          float gv = tanhf_(a2[r] + bg);
          float ov = sigmoidf_(a3[r] + bo);
          float cc = iv * gv;                       // f*0 dropped
          myH[(mrow + r) * HDIM + col] = (half_t)(ov * tanhf_(cc));
        }
      }
    }
    v16h hb[4];
    #pragma unroll
    for (int kb = 0; kb < 4; ++kb)
      hb[kb] = load_a16(myH + row * HDIM + kb * 32, koff);

    // ---- MLP: relu([h_fwd,h_bwd] @ W1 + b1) @ W2 + b2
    float p[8];
    #pragma unroll
    for (int r = 0; r < 8; ++r) p[r] = 0.f;
    #pragma unroll 1
    for (int nb2 = 0; nb2 < 2; ++nb2) {
      v8f f0 = zacc;
      #pragma unroll
      for (int kb = 0; kb < 4; ++kb)
        f0 = wmma16(ha[kb], load_bpack(s_BW1 + (nb2 * 8 + kb) * 512, lane), f0);
      #pragma unroll
      for (int kb = 0; kb < 4; ++kb)
        f0 = wmma16(hb[kb], load_bpack(s_BW1 + (nb2 * 8 + 4 + kb) * 512, lane), f0);
      float bb = b1[nb2 * 16 + row];
      float w2v = W2[nb2 * 16 + row];
      #pragma unroll
      for (int r = 0; r < 8; ++r) {
        float fv = fmaxf(f0[r] + bb, 0.f);
        p[r] += fv * w2v;
      }
    }
    // reduce dot over the 16 N-lanes of each half-wave
    #pragma unroll
    for (int off = 8; off; off >>= 1) {
      #pragma unroll
      for (int r = 0; r < 8; ++r) p[r] += __shfl_xor(p[r], off, 16);
    }
    if (row == 0) {
      float b2v = b2[0];
      int base = e0 + mrow;
      #pragma unroll
      for (int r = 0; r < 8; ++r) ea[base + r] = fmaxf(p[r] + b2v, 0.f);
    }
  }
}

// ------------------------------------------------------- GATv2 stages
__global__ void k_scatter(const float* __restrict__ xorig, const int* __restrict__ ei,
                          const float* __restrict__ ea, float* __restrict__ xmod) {
  int stride = gridDim.x * blockDim.x;
  for (int e = blockIdx.x * blockDim.x + threadIdx.x; e < N_EDGES; e += stride) {
    int s = ei[e], d = ei[N_EDGES + e];
    const float* xs = xorig + (size_t)s * DDIM;
    const float* xd = xorig + (size_t)d * DDIM;
    int sa = 0, da = 0; float sm = xs[0], dm = xd[0];
    for (int j = 1; j < SID_N; ++j) {
      float v = xs[j]; if (v > sm) { sm = v; sa = j; }
      float w = xd[j]; if (w > dm) { dm = w; da = j; }
    }
    float v = ea[e];
    xmod[(size_t)d * DDIM + IN_IDX_C + sa] = v;
    xmod[(size_t)s * DDIM + OUT_IDX_C + da] = v;
  }
}

__global__ void k_pad_half(const float* __restrict__ x, half_t* __restrict__ xh) {
  long long n = (long long)N_NODES * DPAD;
  long long stride = (long long)gridDim.x * blockDim.x;
  for (long long i = (long long)blockIdx.x * blockDim.x + threadIdx.x; i < n; i += stride) {
    int r = (int)(i / DPAD), c = (int)(i % DPAD);
    xh[i] = (half_t)((c < DDIM) ? x[(long long)r * DDIM + c] : 0.f);
  }
}

// wave-per-16x16-tile GEMM: out[N,373] = xh[N,384] @ Bpack(384x384 f16) + bias
__global__ void k_gat_gemm(const half_t* __restrict__ xh, const half_t* __restrict__ Bp,
                           const float* __restrict__ bias, float* __restrict__ out) {
  const int lane = threadIdx.x & 31;
  const int wave = threadIdx.x >> 5;
  const int nw = blockDim.x >> 5;
  const int row = lane & 15;
  const int koff = (lane & 16) ? 8 : 0;
  const int mrow = (lane & 16) ? 8 : 0;
  const int MT = N_NODES / 16, NT = DPAD / 16, KT = DPAD / 32;
  int wg = blockIdx.x * nw + wave;
  int ws = gridDim.x * nw;
  #pragma unroll 1
  for (int tIdx = wg; tIdx < MT * NT; tIdx += ws) {
    int mt = tIdx / NT, nt = tIdx - mt * NT;
    v8f acc;
    #pragma unroll
    for (int r = 0; r < 8; ++r) acc[r] = 0.f;
    const half_t* arow = xh + (size_t)(mt * 16 + row) * DPAD;
    const half_t* btile = Bp + (size_t)nt * KT * 512;
    #pragma unroll 1
    for (int kb = 0; kb < KT; ++kb) {
      __builtin_prefetch(btile + (kb + 1) * 512 + lane * 16, 0, 1);
      union { v16h v; v8h h[2]; } a;
      a.h[0] = *(const v8h*)(arow + kb * 32 + koff);
      a.h[1] = *(const v8h*)(arow + kb * 32 + koff + 16);
      v16h b = load_bpack(btile + kb * 512, lane);
      acc = wmma16(a.v, b, acc);
    }
    int col = nt * 16 + row;
    if (col < DDIM) {
      float bb = bias[col];
      int m0 = mt * 16 + mrow;
      #pragma unroll
      for (int r = 0; r < 8; ++r) out[(size_t)(m0 + r) * DDIM + col] = acc[r] + bb;
    }
  }
}

__global__ void k_edge_e(const int* __restrict__ ei, const float* __restrict__ xl,
                         const float* __restrict__ xr, const float* __restrict__ att,
                         float* __restrict__ evec, float* __restrict__ emax) {
  int lane = threadIdx.x & 31;
  int wid = (blockIdx.x * blockDim.x + threadIdx.x) >> 5;
  int wstride = (gridDim.x * blockDim.x) >> 5;
  for (int e = wid; e < N_EDGES; e += wstride) {
    int s = ei[e], d = ei[N_EDGES + e];
    const float* ps = xl + (size_t)s * DDIM;
    const float* pd = xr + (size_t)d * DDIM;
    float sum = 0.f;
    for (int j = lane; j < DDIM; j += 32) {
      float m = ps[j] + pd[j];
      m = (m > 0.f) ? m : NSLOPE * m;
      sum += m * att[j];
    }
    #pragma unroll
    for (int off = 16; off; off >>= 1) sum += __shfl_xor(sum, off, 32);
    if (lane == 0) { evec[e] = sum; atomic_max_f32(emax + d, sum); }
  }
}

__global__ void k_edge_softmax(const int* __restrict__ ei, const float* __restrict__ evec,
                               const float* __restrict__ emax, float* __restrict__ exv,
                               float* __restrict__ denom) {
  int stride = gridDim.x * blockDim.x;
  for (int e = blockIdx.x * blockDim.x + threadIdx.x; e < N_EDGES; e += stride) {
    int d = ei[N_EDGES + e];
    float v = __expf(evec[e] - emax[d]);
    exv[e] = v;
    atomicAdd(denom + d, v);
  }
}

__global__ void k_edge_agg(const int* __restrict__ ei, const float* __restrict__ exv,
                           const float* __restrict__ denom, const float* __restrict__ xl,
                           float* __restrict__ out, int colOff) {
  int lane = threadIdx.x & 31;
  int wid = (blockIdx.x * blockDim.x + threadIdx.x) >> 5;
  int wstride = (gridDim.x * blockDim.x) >> 5;
  for (int e = wid; e < N_EDGES; e += wstride) {
    int s = ei[e], d = ei[N_EDGES + e];
    float alpha = exv[e] / (denom[d] + 1e-16f);
    const float* ps = xl + (size_t)s * DDIM;
    float* po = out + (size_t)d * OUT_STRIDE + colOff;
    for (int j = lane; j < DDIM; j += 32) atomicAdd(po + j, alpha * ps[j]);
  }
}

__global__ void k_final(const float* __restrict__ xmod, const float* __restrict__ bias,
                        float* __restrict__ out, int colOff) {
  long long n = (long long)N_NODES * DDIM;
  long long stride = (long long)gridDim.x * blockDim.x;
  for (long long i = (long long)blockIdx.x * blockDim.x + threadIdx.x; i < n; i += stride) {
    int r = (int)(i / DDIM), c = (int)(i % DDIM);
    float* o = out + (size_t)r * OUT_STRIDE + colOff + c;
    *o = fmaxf(*o + bias[c] + xmod[i], 0.f);
  }
}

// ---------------------------------------------------------------- launch
extern "C" void kernel_launch(void* const* d_in, const int* in_sizes, int n_in,
                              void* d_out, int out_size, void* d_ws, size_t ws_size,
                              hipStream_t stream) {
  const float* fx    = (const float*)d_in[0];
  const float* bx    = (const float*)d_in[1];
  const int*   fei   = (const int*)d_in[2];
  const int*   bei   = (const int*)d_in[3];
  const int*   ftok  = (const int*)d_in[4];
  const int*   btok  = (const int*)d_in[5];
  const float* embed = (const float*)d_in[6];
  const float* Wih_f = (const float*)d_in[7];
  const float* Whh_f = (const float*)d_in[8];
  const float* bih_f = (const float*)d_in[9];
  const float* bhh_f = (const float*)d_in[10];
  const float* Wih_r = (const float*)d_in[11];
  const float* Whh_r = (const float*)d_in[12];
  const float* bih_r = (const float*)d_in[13];
  const float* bhh_r = (const float*)d_in[14];
  const float* W1    = (const float*)d_in[15];
  const float* b1    = (const float*)d_in[16];
  const float* W2    = (const float*)d_in[17];
  const float* b2    = (const float*)d_in[18];
  float* out = (float*)d_out;

  char* ws = (char*)d_ws;
  size_t off = 0;
  auto alloc = [&](size_t bytes) -> void* {
    void* p = ws + off;
    off += (bytes + 255) & ~(size_t)255;
    return p;
  };
  // NOTE: the first seven allocations are all 256B multiples -> contiguous
  // weight image of WIMG_BYTES, async-copied to LDS by k_lstm_edges.
  half_t* embed_h = (half_t*)alloc(2048 * 2);
  half_t* BxF  = (half_t*)alloc((size_t)32  * 512 * 2);
  half_t* BhF  = (half_t*)alloc((size_t)128 * 512 * 2);
  half_t* BxR  = (half_t*)alloc((size_t)32  * 512 * 2);
  half_t* BW1p = (half_t*)alloc((size_t)16  * 512 * 2);
  float* bfv   = (float*)alloc(512 * 4);
  float* brv   = (float*)alloc(512 * 4);
  half_t* WlP  = (half_t*)alloc((size_t)288 * 512 * 2);   // 24 nb x 12 kb
  half_t* WrP  = (half_t*)alloc((size_t)288 * 512 * 2);
  float* f_ea  = (float*)alloc((size_t)N_EDGES * 4);
  float* b_ea  = (float*)alloc((size_t)N_EDGES * 4);
  float* xmod  = (float*)alloc((size_t)N_NODES * DDIM * 4);
  half_t* xh   = (half_t*)alloc((size_t)N_NODES * DPAD * 2);
  float* xlb   = (float*)alloc((size_t)N_NODES * DDIM * 4);
  float* xrb   = (float*)alloc((size_t)N_NODES * DDIM * 4);
  float* evec  = (float*)alloc((size_t)N_EDGES * 4);
  float* exv   = (float*)alloc((size_t)N_EDGES * 4);
  float* emax  = (float*)alloc((size_t)N_NODES * 4);
  float* denom = (float*)alloc((size_t)N_NODES * 4);

  // output accumulates attention aggregation -> zero it first
  k_fill_f32<<<4096, 256, 0, stream>>>(out, (long long)N_NODES * OUT_STRIDE, 0.f);

  // weight prep
  k_prep_misc<<<8, 256, 0, stream>>>(embed, bih_f, bhh_f, bih_r, bhh_r, embed_h, bfv, brv);
  k_pack_b<<<64, 256, 0, stream>>>(Wih_f, BxF, 32, 512, 32, 1, 1, 32);
  k_pack_b<<<256, 256, 0, stream>>>(Whh_f, BhF, 128, 512, 128, 1, 4, 32);
  k_pack_b<<<64, 256, 0, stream>>>(Wih_r, BxR, 32, 512, 32, 1, 1, 32);
  k_pack_b<<<32, 256, 0, stream>>>(W1, BW1p, 256, 32, 32, 0, 8, 2);

  // edge LSTM encoders (WMMA-heavy; weights resident in LDS via async loads)
  k_lstm_edges<<<640, 128, LSTM_SMEM, stream>>>(ftok, embed_h, b1, W2, b2, f_ea);
  k_lstm_edges<<<640, 128, LSTM_SMEM, stream>>>(btok, embed_h, b1, W2, b2, b_ea);

  // two independent GATv2 passes, sequential so buffers can be reused
  for (int g = 0; g < 2; ++g) {
    const float* x0  = g ? bx : fx;
    const int*   ei  = g ? bei : fei;
    const float* ea  = g ? b_ea : f_ea;
    const float* Wl  = (const float*)d_in[g ? 25 : 19];
    const float* bl  = (const float*)d_in[g ? 26 : 20];
    const float* Wr  = (const float*)d_in[g ? 27 : 21];
    const float* br  = (const float*)d_in[g ? 28 : 22];
    const float* att = (const float*)d_in[g ? 29 : 23];
    const float* gb  = (const float*)d_in[g ? 30 : 24];
    int colOff = g ? DDIM : 0;

    k_pack_b<<<576, 256, 0, stream>>>(Wl, WlP, DDIM, DDIM, DDIM, 0, 12, 24);
    k_pack_b<<<576, 256, 0, stream>>>(Wr, WrP, DDIM, DDIM, DDIM, 0, 12, 24);

    k_copy_f32<<<4096, 256, 0, stream>>>(x0, xmod, (long long)N_NODES * DDIM);
    k_scatter<<<1563, 256, 0, stream>>>(x0, ei, ea, xmod);
    k_pad_half<<<4096, 256, 0, stream>>>(xmod, xh);

    k_fill_f32<<<256, 256, 0, stream>>>(emax, N_NODES, -FLT_MAX);
    k_fill_f32<<<256, 256, 0, stream>>>(denom, N_NODES, 0.f);

    k_gat_gemm<<<2048, 256, 0, stream>>>(xh, WlP, bl, xlb);
    k_gat_gemm<<<2048, 256, 0, stream>>>(xh, WrP, br, xrb);

    k_edge_e<<<2048, 256, 0, stream>>>(ei, xlb, xrb, att, evec, emax);
    k_edge_softmax<<<1563, 256, 0, stream>>>(ei, evec, emax, exv, denom);
    k_edge_agg<<<2048, 256, 0, stream>>>(ei, exv, denom, xlb, out, colOff);
    k_final<<<4096, 256, 0, stream>>>(xmod, gb, out, colOff);
  }
  (void)in_sizes; (void)n_in; (void)out_size; (void)ws_size;
}
